// Comm_OUT_43052752175494
// MI455X (gfx1250) — matrix operand order repaired and checked by reading, not compiled, not run on hardware
//
#include <hip/hip_runtime.h>
#include <hip/hip_bf16.h>

// ---------------------------------------------------------------------------
// Problem dims
// ---------------------------------------------------------------------------
#define EE   64
#define SS   32
#define LL   32
#define HH   1024
#define INN  2048
#define OUTN 64
#define NN   (EE * SS)        // 2048 sequences
#define MM2  (NN * LL)        // 65536 rows for conv / out GEMMs

typedef __attribute__((ext_vector_type(16))) __bf16 v16bf;
typedef __attribute__((ext_vector_type(8)))  __bf16 v8bf;
typedef __attribute__((ext_vector_type(4)))  __bf16 v4bf;
typedef __attribute__((ext_vector_type(8)))  float  v8f;

// ---------------------------------------------------------------------------
// WMMA helpers (CDNA5 16x16x32 bf16, f32 accumulate)
// ---------------------------------------------------------------------------
__device__ __forceinline__ v8f wmma_bf16(v16bf a, v16bf b, v8f c) {
  return __builtin_amdgcn_wmma_f32_16x16x32_bf16(false, a, false, b, (short)0, c,
                                                 false, false);
}

// A-operand loader. Row-major bf16 source. Per ISA layout for 16-bit A 16x32:
// lanes 0-15 hold K = {k0..k0+7, k0+16..k0+23}, lanes 16-31 hold
// K = {k0+8..k0+15, k0+24..k0+31}; v16bf elems 0..7 = low chunk, 8..15 = high.
// `row` must already include this lane's (lane&15) row offset.
__device__ __forceinline__ v16bf wmma_load_a(const __bf16* src, long row,
                                             long rowStride, int k0, int lane,
                                             bool valid) {
  v8bf lo = {}, hi = {};
  if (valid) {
    const __bf16* p = src + row * rowStride + k0 + ((lane >> 4) << 3);
    lo = *(const v8bf*)(p);
    hi = *(const v8bf*)(p + 16);
  }
  return __builtin_shufflevector(lo, hi, 0, 1, 2, 3, 4, 5, 6, 7,
                                 8, 9, 10, 11, 12, 13, 14, 15);
}

// Packed-B tile load: one contiguous 32-byte read per lane.
__device__ __forceinline__ v16bf wmma_load_b(const __bf16* packed, long kt,
                                             long nt, long ntiles, int lane) {
  return *(const v16bf*)(packed + (((kt * ntiles + nt) * 32) + lane) * 16);
}

// 4 A tiles (rows precomputed per lane, optional validity), 4 B tiles, 16 WMMAs.
__device__ __forceinline__ void load_a4(v16bf a[4], const __bf16* src,
                                        const long row[4], long rowStride,
                                        int k0, int lane, const bool val[4]) {
#pragma unroll
  for (int mi = 0; mi < 4; ++mi)
    a[mi] = wmma_load_a(src, row[mi], rowStride, k0, lane, val[mi]);
}
__device__ __forceinline__ void load_b4(v16bf b[4], const __bf16* packed, long kt,
                                        long nt0, long ntiles, int lane) {
#pragma unroll
  for (int j = 0; j < 4; ++j)
    b[j] = wmma_load_b(packed, kt, nt0 + j, ntiles, lane);
}
__device__ __forceinline__ void wmma16(v8f acc[4][4], const v16bf a[4],
                                       const v16bf b[4]) {
#pragma unroll
  for (int mi = 0; mi < 4; ++mi)
#pragma unroll
    for (int j = 0; j < 4; ++j)
      acc[mi][j] = wmma_bf16(a[mi], b[j], acc[mi][j]);
}

// ---------------------------------------------------------------------------
// Prep kernels
// ---------------------------------------------------------------------------
__global__ void k_cvt_bf16(const float* __restrict__ src, __bf16* __restrict__ dst,
                           long n) {
  long i = (long)blockIdx.x * blockDim.x + threadIdx.x;
  if (i < n) dst[i] = (__bf16)src[i];
}

// Pack f32 weight B[k][n] (element at k*sk + n*sn + off) into WMMA B layout:
// linear idx = ((kt*ntiles+nt)*32 + lane)*16 + j,
// k = kt*32 + 8*(lane>=16) + (j&7) + 16*(j>=8), n = nt*16 + (lane&15).
__global__ void k_pack_b(const float* __restrict__ src, __bf16* __restrict__ dst,
                         int K, int N, long sk, long sn, long off) {
  long i = (long)blockIdx.x * blockDim.x + threadIdx.x;
  long total = (long)K * N;
  if (i >= total) return;
  int  j    = (int)(i & 15);
  int  lane = (int)((i >> 4) & 31);
  long tile = i >> 9;
  long ntiles = N >> 4;
  long kt = tile / ntiles;
  long nt = tile % ntiles;
  long k = kt * 32 + ((lane >> 4) << 3) + (j & 7) + ((j >> 3) << 4);
  long n = nt * 16 + (lane & 15);
  dst[i] = (__bf16)src[k * sk + n * sn + off];
}

__global__ void k_zero_f32(float* __restrict__ p, long n) {
  long i = (long)blockIdx.x * blockDim.x + threadIdx.x;
  if (i < n) p[i] = 0.0f;
}

// ---------------------------------------------------------------------------
// GEMM 1: x_r = A0(bf16 2048x2048) @ Wx_packed + bx -> f32 [2048,1024]
// 4 M-tiles per wave (64x64), copy-free ping-pong double buffering.
// grid (2048/64, 2), block 256 (8 waves).
// ---------------------------------------------------------------------------
__global__ __launch_bounds__(256) void k_gemm_xr(const __bf16* __restrict__ A,
                                                 const __bf16* __restrict__ Bp,
                                                 const float* __restrict__ bx,
                                                 float* __restrict__ xr) {
  int lane = threadIdx.x & 31;
  int wave = threadIdx.x >> 5;
  long m0 = (long)blockIdx.x * 64;
  int  n0 = blockIdx.y * 512 + wave * 64;
  const int KT = INN / 32;            // 64 (even)
  long nt0 = n0 >> 4;

  long row[4];
  bool val[4] = {true, true, true, true};
#pragma unroll
  for (int mi = 0; mi < 4; ++mi) row[mi] = m0 + mi * 16 + (lane & 15);

  v8f acc[4][4] = {};
  v16bf a0[4], b0[4], a1[4], b1[4];
  load_a4(a0, A, row, INN, 0, lane, val);
  load_b4(b0, Bp, 0, nt0, HH / 16, lane);

  for (int kt = 0; kt < KT; kt += 2) {
    int k2 = (kt + 2 < KT) ? kt + 2 : kt;   // clamped prefetch
    load_a4(a1, A, row, INN, (kt + 1) * 32, lane, val);
    load_b4(b1, Bp, kt + 1, nt0, HH / 16, lane);
    wmma16(acc, a0, b0);
    load_a4(a0, A, row, INN, k2 * 32, lane, val);
    load_b4(b0, Bp, k2, nt0, HH / 16, lane);
    wmma16(acc, a1, b1);
  }

  int col = lane & 15, rb = (lane >> 4) * 8;
#pragma unroll
  for (int mi = 0; mi < 4; ++mi)
#pragma unroll
    for (int j = 0; j < 4; ++j) {
      int n = n0 + j * 16 + col;
      float bv = bx[n];
#pragma unroll
      for (int r = 0; r < 8; ++r)
        xr[(m0 + mi * 16 + rb + r) * HH + n] = acc[mi][j][r] + bv;
    }
}

// ---------------------------------------------------------------------------
// MTRNN step 0 (h_prev = 0): h = 0.5*tanh(x_r + bh); also hs[:,0,:] (bf16)
// ---------------------------------------------------------------------------
__global__ void k_rnn_init(const float* __restrict__ xr, const float* __restrict__ bh,
                           float* __restrict__ h, __bf16* __restrict__ hs) {
  long i = (long)blockIdx.x * blockDim.x + threadIdx.x;
  if (i >= (long)NN * HH) return;
  int  n   = (int)(i & (HH - 1));
  long row = i >> 10;
  float v = 0.5f * tanhf(xr[i] + bh[n]);
  h[i] = v;
  hs[(row * LL + 0) * HH + n] = (__bf16)v;
}

// ---------------------------------------------------------------------------
// MTRNN step t>=1: h = 0.5*h + 0.5*tanh(x_r + hs[:,t-1,:]@Wh + bh)
// A rows at hs + (m*LL + t-1)*HH => src = hs + (t-1)*HH, stride LL*HH.
// 4 M-tiles per wave, ping-pong; grid (2048/64, 2) block 256.
// ---------------------------------------------------------------------------
__global__ __launch_bounds__(256) void k_rnn_step(const __bf16* __restrict__ hs_c,
                                                  __bf16* __restrict__ hs,
                                                  const __bf16* __restrict__ Whp,
                                                  const float* __restrict__ xr,
                                                  const float* __restrict__ bh,
                                                  float* __restrict__ h, int t) {
  int lane = threadIdx.x & 31;
  int wave = threadIdx.x >> 5;
  long m0 = (long)blockIdx.x * 64;
  int  n0 = blockIdx.y * 512 + wave * 64;
  const __bf16* asrc = hs_c + (long)(t - 1) * HH;
  const int KT = HH / 32;             // 32 (even)
  long nt0 = n0 >> 4;

  long row[4];
  bool val[4] = {true, true, true, true};
#pragma unroll
  for (int mi = 0; mi < 4; ++mi) row[mi] = m0 + mi * 16 + (lane & 15);

  v8f acc[4][4] = {};
  v16bf a0[4], b0[4], a1[4], b1[4];
  load_a4(a0, asrc, row, (long)LL * HH, 0, lane, val);
  load_b4(b0, Whp, 0, nt0, HH / 16, lane);

  for (int kt = 0; kt < KT; kt += 2) {
    int k2 = (kt + 2 < KT) ? kt + 2 : kt;
    load_a4(a1, asrc, row, (long)LL * HH, (kt + 1) * 32, lane, val);
    load_b4(b1, Whp, kt + 1, nt0, HH / 16, lane);
    wmma16(acc, a0, b0);
    load_a4(a0, asrc, row, (long)LL * HH, k2 * 32, lane, val);
    load_b4(b0, Whp, k2, nt0, HH / 16, lane);
    wmma16(acc, a1, b1);
  }

  int col = lane & 15, rb = (lane >> 4) * 8;
#pragma unroll
  for (int mi = 0; mi < 4; ++mi)
#pragma unroll
    for (int j = 0; j < 4; ++j) {
      int n = n0 + j * 16 + col;
      float bv = bh[n];
#pragma unroll
      for (int r = 0; r < 8; ++r) {
        long m = m0 + mi * 16 + rb + r;
        long idx = m * HH + n;
        float hn = 0.5f * h[idx] + 0.5f * tanhf(xr[idx] + bv + acc[mi][j][r]);
        h[idx] = hn;
        hs[(m * LL + t) * HH + n] = (__bf16)hn;
      }
    }
}

// ---------------------------------------------------------------------------
// Multi-kernel conv over L as tap-shifted GEMMs.
// Wave = 64 rows x 64 cols (4 M-tiles, B reused across them), ping-pong
// within each tap. Branch = channel block of 256 -> kernel size 1/3/5/7;
// tap shift d moves the hs row by d (valid iff 0 <= l+d < 32).
// Stores y (pre-BN) as bf16 [65536,1024]. grid (65536/64, 2), block 256.
// ---------------------------------------------------------------------------
__global__ __launch_bounds__(256) void k_conv(const __bf16* __restrict__ hs,
                                              const __bf16* __restrict__ Wc,
                                              const float* __restrict__ b1,
                                              const float* __restrict__ b3,
                                              const float* __restrict__ b5,
                                              const float* __restrict__ b7,
                                              __bf16* __restrict__ y) {
  int lane = threadIdx.x & 31;
  int wave = threadIdx.x >> 5;
  long m0 = (long)blockIdx.x * 64;
  int  n0 = blockIdx.y * 512 + wave * 64;
  int branch = n0 >> 8;                 // 0..3
  int ksz = 1 + 2 * branch;
  int pad = branch;
  const int prefix[4] = {0, 1, 4, 9};   // taps before branch
  const __bf16* Wb = Wc + (long)prefix[branch] * (1024L * 256);
  long ntLocal = (n0 - branch * 256) >> 4;
  const int KT = HH / 32;               // 32 (even)

  long mrow[4];
  int  lidx[4];
#pragma unroll
  for (int mi = 0; mi < 4; ++mi) {
    mrow[mi] = m0 + mi * 16 + (lane & 15);
    lidx[mi] = (int)(mrow[mi] & (LL - 1));
  }

  v8f acc[4][4] = {};
  for (int ti = 0; ti < ksz; ++ti) {
    int d = ti - pad;
    long row[4];
    bool val[4];
#pragma unroll
    for (int mi = 0; mi < 4; ++mi) {
      row[mi] = mrow[mi] + d;
      val[mi] = ((unsigned)(lidx[mi] + d)) < (unsigned)LL;
    }
    const __bf16* tapB = Wb + (long)ti * (1024L * 256);

    v16bf a0[4], b0[4], a1[4], b1[4];
    load_a4(a0, hs, row, HH, 0, lane, val);
    load_b4(b0, tapB, 0, ntLocal, 256 / 16, lane);
    for (int kt = 0; kt < KT; kt += 2) {
      int k2 = (kt + 2 < KT) ? kt + 2 : kt;
      load_a4(a1, hs, row, HH, (kt + 1) * 32, lane, val);
      load_b4(b1, tapB, kt + 1, ntLocal, 256 / 16, lane);
      wmma16(acc, a0, b0);
      load_a4(a0, hs, row, HH, k2 * 32, lane, val);
      load_b4(b0, tapB, k2, ntLocal, 256 / 16, lane);
      wmma16(acc, a1, b1);
    }
  }

  const float* bb = (branch == 0) ? b1 : (branch == 1) ? b3 : (branch == 2) ? b5 : b7;
  int col = lane & 15, rb = (lane >> 4) * 8;
#pragma unroll
  for (int mi = 0; mi < 4; ++mi)
#pragma unroll
    for (int j = 0; j < 4; ++j) {
      int n = n0 + j * 16 + col;
      float bv = bb[n - branch * 256];
#pragma unroll
      for (int r = 0; r < 8; ++r)
        y[(m0 + mi * 16 + rb + r) * HH + n] = (__bf16)(acc[mi][j][r] + bv);
    }
}

// ---------------------------------------------------------------------------
// BN batch stats: per-channel sum & sumsq over 65536 rows.
// ---------------------------------------------------------------------------
__global__ __launch_bounds__(256) void k_stats(const __bf16* __restrict__ y,
                                               float* __restrict__ stats) {
  int t = threadIdx.x;
  int c0 = t * 4;
  long row0 = (long)blockIdx.x * (MM2 / 256);
  float s0 = 0, s1 = 0, s2 = 0, s3 = 0;
  float q0 = 0, q1 = 0, q2 = 0, q3 = 0;
  for (int r = 0; r < MM2 / 256; ++r) {
    v4bf v = *(const v4bf*)(y + (row0 + r) * HH + c0);
    float a = (float)v[0], b = (float)v[1], c = (float)v[2], d = (float)v[3];
    s0 += a; s1 += b; s2 += c; s3 += d;
    q0 += a * a; q1 += b * b; q2 += c * c; q3 += d * d;
  }
  atomicAdd(&stats[c0 + 0], s0); atomicAdd(&stats[c0 + 1], s1);
  atomicAdd(&stats[c0 + 2], s2); atomicAdd(&stats[c0 + 3], s3);
  atomicAdd(&stats[HH + c0 + 0], q0); atomicAdd(&stats[HH + c0 + 1], q1);
  atomicAdd(&stats[HH + c0 + 2], q2); atomicAdd(&stats[HH + c0 + 3], q3);
}

__global__ void k_bnprep(const float* __restrict__ stats,
                         const float* __restrict__ gamma,
                         const float* __restrict__ beta,
                         float* __restrict__ ss) {
  int c = blockIdx.x * blockDim.x + threadIdx.x;
  if (c >= HH) return;
  const float inv = 1.0f / (float)MM2;
  float mean = stats[c] * inv;
  float var  = stats[HH + c] * inv - mean * mean;
  float sc = gamma[c] * rsqrtf(var + 1e-5f);
  ss[c] = sc;
  ss[HH + c] = beta[c] - mean * sc;
}

// ---------------------------------------------------------------------------
// Output GEMM: out[m,0:64] = PReLU(BN(y[m,:])) @ Wout + bout, f32 output.
// Wave = 64 rows x 64 cols (4 M-tiles share the 4 B tiles covering all OUT).
// BN affine + PReLU fused into the A load (scale/shift staged in LDS); the
// BN VALU work fills the load shadow, so no manual pipelining here.
// grid 65536/64/8 = 128 blocks, block 256 (8 waves).
// ---------------------------------------------------------------------------
__global__ __launch_bounds__(256) void k_out(const __bf16* __restrict__ y,
                                             const __bf16* __restrict__ Wop,
                                             const float* __restrict__ ss,
                                             const float* __restrict__ prelu_a,
                                             const float* __restrict__ bout,
                                             float* __restrict__ out) {
  __shared__ float s_scale[HH];
  __shared__ float s_shift[HH];
  for (int i = threadIdx.x; i < HH; i += blockDim.x) {
    s_scale[i] = ss[i];
    s_shift[i] = ss[HH + i];
  }
  __syncthreads();
  float pa = prelu_a[0];

  int lane = threadIdx.x & 31;
  int wave = threadIdx.x >> 5;
  long m0 = ((long)blockIdx.x * 8 + wave) * 64;

  v8f acc[4][4] = {};
  for (int kt = 0; kt < HH / 32; ++kt) {
    int kb = kt * 32 + ((lane >> 4) << 3);
    v16bf b[4], a[4];
#pragma unroll
    for (int j = 0; j < 4; ++j)
      b[j] = wmma_load_b(Wop, kt, j, OUTN / 16, lane);
#pragma unroll
    for (int mi = 0; mi < 4; ++mi) {
      const __bf16* p = y + (m0 + mi * 16 + (lane & 15)) * HH + kb;
      v8bf lo = *(const v8bf*)(p);
      v8bf hi = *(const v8bf*)(p + 16);
#pragma unroll
      for (int i = 0; i < 8; ++i) {
        float v = (float)lo[i];
        v = v * s_scale[kb + i] + s_shift[kb + i];
        v = (v > 0.0f) ? v : pa * v;
        a[mi][i] = (__bf16)v;
        float w = (float)hi[i];
        w = w * s_scale[kb + 16 + i] + s_shift[kb + 16 + i];
        w = (w > 0.0f) ? w : pa * w;
        a[mi][8 + i] = (__bf16)w;
      }
    }
#pragma unroll
    for (int mi = 0; mi < 4; ++mi)
#pragma unroll
      for (int j = 0; j < 4; ++j)
        acc[mi][j] = wmma_bf16(a[mi], b[j], acc[mi][j]);
  }

  int col = lane & 15, rb = (lane >> 4) * 8;
#pragma unroll
  for (int mi = 0; mi < 4; ++mi)
#pragma unroll
    for (int j = 0; j < 4; ++j) {
      int o = j * 16 + col;
      float bv = bout[o];
#pragma unroll
      for (int r = 0; r < 8; ++r)
        out[(m0 + mi * 16 + rb + r) * OUTN + o] = acc[mi][j][r] + bv;
    }
}

// ---------------------------------------------------------------------------
// Orchestration
// ---------------------------------------------------------------------------
extern "C" void kernel_launch(void* const* d_in, const int* in_sizes, int n_in,
                              void* d_out, int out_size, void* d_ws, size_t ws_size,
                              hipStream_t stream) {
  const float* h_w_action = (const float*)d_in[0];
  const float* Wx   = (const float*)d_in[1];
  const float* bx   = (const float*)d_in[2];
  const float* Wh   = (const float*)d_in[3];
  const float* bh   = (const float*)d_in[4];
  const float* w1   = (const float*)d_in[5];
  const float* b1   = (const float*)d_in[6];
  const float* w3   = (const float*)d_in[7];
  const float* b3   = (const float*)d_in[8];
  const float* w5   = (const float*)d_in[9];
  const float* b5   = (const float*)d_in[10];
  const float* w7   = (const float*)d_in[11];
  const float* b7   = (const float*)d_in[12];
  const float* gamma   = (const float*)d_in[13];
  const float* beta    = (const float*)d_in[14];
  const float* prelu_a = (const float*)d_in[15];
  const float* Wout    = (const float*)d_in[16];
  const float* bout    = (const float*)d_in[17];
  float* out = (float*)d_out;

  // Workspace carve (all 256B-aligned). Total ~308 MB.
  char* base = (char*)d_ws;
  size_t off = 0;
  auto carve = [&](size_t bytes) -> char* {
    char* p = base + off;
    off = (off + bytes + 255) & ~(size_t)255;
    return p;
  };
  __bf16* ws_a0   = (__bf16*)carve((size_t)NN * INN * 2);
  __bf16* ws_wxp  = (__bf16*)carve((size_t)INN * HH * 2);
  __bf16* ws_whp  = (__bf16*)carve((size_t)HH * HH * 2);
  __bf16* ws_wcp  = (__bf16*)carve((size_t)16 * 1024 * 256 * 2);
  __bf16* ws_wop  = (__bf16*)carve((size_t)HH * OUTN * 2);
  float*  ws_xr   = (float*) carve((size_t)NN * HH * 4);
  float*  ws_h    = (float*) carve((size_t)NN * HH * 4);
  __bf16* ws_hs   = (__bf16*)carve((size_t)NN * LL * HH * 2);
  __bf16* ws_y    = (__bf16*)carve((size_t)MM2 * HH * 2);
  float*  ws_st   = (float*) carve((size_t)2 * HH * 4);
  float*  ws_ss   = (float*) carve((size_t)2 * HH * 4);

  // --- prep: zero stats, convert A0, pack weights ---
  k_zero_f32<<<(2 * HH + 255) / 256, 256, 0, stream>>>(ws_st, 2 * HH);
  {
    long n = (long)NN * INN;
    k_cvt_bf16<<<(unsigned)((n + 255) / 256), 256, 0, stream>>>(h_w_action, ws_a0, n);
  }
  k_pack_b<<<(INN * HH + 255) / 256, 256, 0, stream>>>(Wx, ws_wxp, INN, HH, HH, 1, 0);
  k_pack_b<<<(HH * HH + 255) / 256, 256, 0, stream>>>(Wh, ws_whp, HH, HH, HH, 1, 0);
  {
    const float* wsrc[4] = {w1, w3, w5, w7};
    long slab = 1024L * 256;
    int tapbase = 0;
    for (int b = 0; b < 4; ++b) {
      int kw = 1 + 2 * b;
      for (int t = 0; t < kw; ++t) {
        k_pack_b<<<(1024 * 256 + 255) / 256, 256, 0, stream>>>(
            wsrc[b], ws_wcp + (long)(tapbase + t) * slab,
            1024, 256, /*sk=*/kw, /*sn=*/(long)HH * kw, /*off=*/t);
      }
      tapbase += kw;
    }
  }
  k_pack_b<<<(HH * OUTN + 255) / 256, 256, 0, stream>>>(Wout, ws_wop, HH, OUTN, OUTN, 1, 0);

  // --- x_r GEMM ---
  k_gemm_xr<<<dim3(NN / 64, 2), 256, 0, stream>>>(ws_a0, ws_wxp, bx, ws_xr);

  // --- MTRNN recurrence ---
  {
    long n = (long)NN * HH;
    k_rnn_init<<<(unsigned)((n + 255) / 256), 256, 0, stream>>>(ws_xr, bh, ws_h, ws_hs);
  }
  for (int t = 1; t < LL; ++t)
    k_rnn_step<<<dim3(NN / 64, 2), 256, 0, stream>>>(ws_hs, ws_hs, ws_whp, ws_xr,
                                                     bh, ws_h, t);

  // --- conv (tap-shifted WMMA GEMMs) ---
  k_conv<<<dim3(MM2 / 64, 2), 256, 0, stream>>>(ws_hs, ws_wcp, b1, b3, b5, b7, ws_y);

  // --- BN stats + finalize ---
  k_stats<<<256, 256, 0, stream>>>(ws_y, ws_st);
  k_bnprep<<<HH / 256, 256, 0, stream>>>(ws_st, gamma, beta, ws_ss);

  // --- fused BN/PReLU + output GEMM ---
  k_out<<<MM2 / 64 / 8, 256, 0, stream>>>(ws_y, ws_wop, ws_ss, prelu_a, bout, out);

  (void)in_sizes; (void)n_in; (void)out_size; (void)ws_size;
}
